// MultiHeadAttention_66383014527497
// MI455X (gfx1250) — compile-verified, matrix operand order checked by reading, not run on hardware
//
#include <hip/hip_runtime.h>

typedef __attribute__((ext_vector_type(16))) __bf16 v16bf;
typedef __attribute__((ext_vector_type(8)))  float  v8f;

#define EMBED 1024
#define NHEAD 16
#define HDIM  64
#define SEQ   1024
#define BATCH 8
#define N3    3072
#define NTOK  (BATCH * SEQ)   // 8192

union Frag {
    v16bf v;
    unsigned u[8];
};

// K-offset of VGPR-pair p for the 16-bit WMMA A/B fragment layout (wave32).
// lanes 0-15: p0..3 -> K 0..7,  p4..7 -> K 16..23
// lanes16-31: p0..3 -> K 8..15, p4..7 -> K 24..31
__device__ __forceinline__ int frag_k0(int p, int hi) {
    return ((p & 3) << 1) + (hi << 3) + ((p & 4) << 2);
}

__device__ __forceinline__ void load_frag(Frag& f, const __bf16* base, int hi) {
#pragma unroll
    for (int p = 0; p < 8; ++p)
        f.u[p] = *reinterpret_cast<const unsigned*>(base + frag_k0(p, hi));
}

__device__ __forceinline__ v8f wmma_bf16(const Frag& a, const Frag& b, v8f c) {
    return __builtin_amdgcn_wmma_f32_16x16x32_bf16(
        false, a.v, false, b.v, (short)0, c, false, false);
}

// ---- gfx1250 async global->LDS copy (ASYNCcnt path), 16B per lane ---------
__device__ __forceinline__ unsigned lds_off_u32(const void* p) {
    // generic LDS address: low 32 bits are the byte offset within LDS
    return (unsigned)(unsigned long long)p;
}
__device__ __forceinline__ void async_cp16(unsigned lds_off, const void* g) {
    unsigned long long ga = (unsigned long long)g;
    asm volatile("global_load_async_to_lds_b128 %0, %1, off"
                 :: "v"(lds_off), "v"(ga) : "memory");
}

// ---------------------------------------------------------------- converts
__global__ void __launch_bounds__(256) k_cvt(const float* __restrict__ in,
                                             __bf16* __restrict__ out, int n) {
    int i = blockIdx.x * 256 + threadIdx.x;
    if (i < n) out[i] = (__bf16)in[i];
}

// in: [K,N] row-major f32  ->  out: [N,K] bf16 (transposed)
__global__ void __launch_bounds__(256) k_cvt_t(const float* __restrict__ in,
                                               __bf16* __restrict__ out,
                                               int K, int N) {
    int i = blockIdx.x * 256 + threadIdx.x;
    if (i < K * N) {
        int n = i % N, k = i / N;
        out[n * K + k] = (__bf16)in[i];
    }
}

// ---------------------------------------------------------------- QKV GEMM
// A[NTOK,EMBED] @ Wt[N3,EMBED]^T + bias -> scatter into Qb/Kb (B*H,S,Hd) and
// Vt (B*H,Hd,S).  Softmax scale (1/8) folded into Q.
// 2-deep register double-buffer: next k-step's fragments load while the
// current step's 4 WMMAs execute.
__global__ void __launch_bounds__(256) k_gemm_qkv(
    const __bf16* __restrict__ A, const __bf16* __restrict__ Wt,
    const float* __restrict__ bias,
    __bf16* __restrict__ Qb, __bf16* __restrict__ Kb, __bf16* __restrict__ Vt) {
    const int lane = threadIdx.x & 31;
    const int wave = threadIdx.x >> 5;
    const int hi = lane >> 4, l15 = lane & 15;
    const int tileM = blockIdx.x * 32 + (wave >> 2) * 16;
    const int tileN = blockIdx.y * 256 + (wave & 3) * 64;

    const __bf16* arow = A + (size_t)(tileM + l15) * EMBED;
    const __bf16* bcol[4];
#pragma unroll
    for (int j = 0; j < 4; ++j)
        bcol[j] = Wt + (size_t)(tileN + j * 16 + l15) * EMBED;

    v8f acc[4] = {};
    Frag a0, a1, b0[4], b1[4];
    load_frag(a0, arow, hi);
#pragma unroll
    for (int j = 0; j < 4; ++j) load_frag(b0[j], bcol[j], hi);

#pragma unroll 1
    for (int k = 0; k < EMBED; k += 64) {
        const int k1 = k + 32;
        load_frag(a1, arow + k1, hi);
#pragma unroll
        for (int j = 0; j < 4; ++j) load_frag(b1[j], bcol[j] + k1, hi);
#pragma unroll
        for (int j = 0; j < 4; ++j) acc[j] = wmma_bf16(a0, b0[j], acc[j]);

        const int k2 = (k + 64) & (EMBED - 1);   // wrap: harmless tail reload
        load_frag(a0, arow + k2, hi);
#pragma unroll
        for (int j = 0; j < 4; ++j) load_frag(b0[j], bcol[j] + k2, hi);
#pragma unroll
        for (int j = 0; j < 4; ++j) acc[j] = wmma_bf16(a1, b1[j], acc[j]);
    }

#pragma unroll
    for (int j = 0; j < 4; ++j) {
        int col = tileN + j * 16 + l15;          // 0..3071
        int sel = col >> 10;                     // 0=Q 1=K 2=V
        int rem = col & 1023;
        int h = rem >> 6, hd = rem & 63;
        float bv = bias[col];
#pragma unroll
        for (int r = 0; r < 8; ++r) {
            int row = tileM + r + hi * 8;        // token index
            int b = row >> 10, s = row & 1023;
            int bh = b * NHEAD + h;
            float val = acc[j][r] + bv;
            if (sel == 0)
                Qb[((size_t)bh * SEQ + s) * HDIM + hd] = (__bf16)(val * 0.125f);
            else if (sel == 1)
                Kb[((size_t)bh * SEQ + s) * HDIM + hd] = (__bf16)val;
            else
                Vt[((size_t)bh * HDIM + hd) * SEQ + s] = (__bf16)val;
        }
    }
}

// ---------------------------------------------------------------- attention
// One wave = 16 query rows.  All 4 waves of a workgroup share the same (b,h),
// so each 32-key K tile (32x64, 4KB) and V tile (64x32, 4KB) is staged ONCE
// per workgroup into LDS with async global->LDS copies, double-buffered.
__global__ void __launch_bounds__(128) k_attn(
    const __bf16* __restrict__ Qb, const __bf16* __restrict__ Kb,
    const __bf16* __restrict__ Vt, __bf16* __restrict__ ctx) {
    __shared__ __bf16 ldsK[2][32 * HDIM];   // [key][feat]
    __shared__ __bf16 ldsV[2][HDIM * 32];   // [feat][key]
    __shared__ __bf16 ldsP[4][16 * 32];     // per-wave P staging

    const int tid  = threadIdx.x;
    const int lane = tid & 31;
    const int wave = tid >> 5;
    const int hi = lane >> 4, l15 = lane & 15;
    const int bh = blockIdx.x;
    const int qStart = blockIdx.y * 64 + wave * 16;
    const __bf16* Q = Qb + (size_t)bh * SEQ * HDIM;
    const __bf16* K = Kb + (size_t)bh * SEQ * HDIM;
    const __bf16* V = Vt + (size_t)bh * HDIM * SEQ;
    __bf16* pw = &ldsP[wave][0];

    const unsigned kOff[2] = { lds_off_u32(&ldsK[0][0]), lds_off_u32(&ldsK[1][0]) };
    const unsigned vOff[2] = { lds_off_u32(&ldsV[0][0]), lds_off_u32(&ldsV[1][0]) };

    // cooperative async prefetch of one 32-key K/V tile pair (4 asyncs/thread)
    auto prefetch = [&](int buf, int kS) {
        const char* kg = (const char*)(K + (size_t)kS * HDIM);  // 4KB contiguous
        const char* vg = (const char*)(V + kS);                 // 64 rows x 64B
#pragma unroll
        for (int i = 0; i < 2; ++i) {
            int c = tid + i * 128;              // chunk 0..255 (16B each)
            async_cp16(kOff[buf] + c * 16, kg + c * 16);
            int f = c >> 2, colc = c & 3;
            async_cp16(vOff[buf] + c * 16,
                       vg + (size_t)f * (SEQ * 2) + colc * 16);
        }
    };

    Frag aq0, aq1;
    {
        const __bf16* qrow = Q + (size_t)(qStart + l15) * HDIM;
        load_frag(aq0, qrow, hi);
        load_frag(aq1, qrow + 32, hi);
    }

    float mrow[8], lrow[8];
#pragma unroll
    for (int r = 0; r < 8; ++r) { mrow[r] = -1e30f; lrow[r] = 0.f; }
    v8f acc[4] = {};

    prefetch(0, 0);
    int cur = 0;
#pragma unroll 1
    for (int kS = 0; kS < SEQ; kS += 32) {
        __syncthreads();                         // buf^1 free to overwrite
        prefetch(cur ^ 1, (kS + 32) & (SEQ - 1));
        asm volatile("s_wait_asynccnt 0x4" ::: "memory");  // cur group landed
        __syncthreads();                         // ... for all waves

        // ---- scores: two 16x16 tiles from LDS K, K-dim 64 = 2 chained WMMAs
        v8f sc[2];
#pragma unroll
        for (int t = 0; t < 2; ++t) {
            const __bf16* krow = &ldsK[cur][(t * 16 + l15) * HDIM];
            Frag b0, b1;
            load_frag(b0, krow, hi);
            load_frag(b1, krow + 32, hi);
            v8f c = {};
            c = wmma_bf16(aq0, b0, c);
            c = wmma_bf16(aq1, b1, c);
            sc[t] = c;
        }
        // ---- row max over 32 keys (reduce the 16 lanes holding one row)
        float rmax[8];
#pragma unroll
        for (int r = 0; r < 8; ++r) rmax[r] = fmaxf(sc[0][r], sc[1][r]);
#pragma unroll
        for (int m = 1; m <= 8; m <<= 1)
#pragma unroll
            for (int r = 0; r < 8; ++r)
                rmax[r] = fmaxf(rmax[r], __shfl_xor(rmax[r], m, 32));
        // ---- online softmax update
        float alpha[8], rsum[8];
#pragma unroll
        for (int r = 0; r < 8; ++r) {
            float mn = fmaxf(mrow[r], rmax[r]);
            alpha[r] = __expf(mrow[r] - mn);
            mrow[r] = mn;
            rsum[r] = 0.f;
        }
#pragma unroll
        for (int t = 0; t < 2; ++t)
#pragma unroll
            for (int r = 0; r < 8; ++r) {
                float pv = __expf(sc[t][r] - mrow[r]);
                rsum[r] += pv;
                pw[(r + hi * 8) * 32 + t * 16 + l15] = (__bf16)pv;
            }
#pragma unroll
        for (int m = 1; m <= 8; m <<= 1)
#pragma unroll
            for (int r = 0; r < 8; ++r)
                rsum[r] += __shfl_xor(rsum[r], m, 32);
#pragma unroll
        for (int r = 0; r < 8; ++r) lrow[r] = lrow[r] * alpha[r] + rsum[r];
#pragma unroll
        for (int j = 0; j < 4; ++j)
#pragma unroll
            for (int r = 0; r < 8; ++r) acc[j][r] *= alpha[r];
        // ---- wait for this wave's P writes, reload P as A-fragment
        asm volatile("s_wait_dscnt 0x0" ::: "memory");
        Frag ap;
        load_frag(ap, pw + l15 * 32, hi);
        // ---- P(16x32) @ V(32x64) from LDS V
#pragma unroll
        for (int j = 0; j < 4; ++j) {
            const __bf16* vrow = &ldsV[cur][(j * 16 + l15) * 32];
            Frag bv;
            load_frag(bv, vrow, hi);
            acc[j] = wmma_bf16(ap, bv, acc[j]);
        }
        cur ^= 1;
    }
    // ---- normalize + write context [B,S,H*Hd]
    const int b = bh >> 4, h = bh & 15;
#pragma unroll
    for (int j = 0; j < 4; ++j) {
        int col = h * HDIM + j * 16 + l15;
#pragma unroll
        for (int r = 0; r < 8; ++r) {
            int tok = b * SEQ + qStart + r + hi * 8;
            ctx[(size_t)tok * EMBED + col] = (__bf16)(acc[j][r] / lrow[r]);
        }
    }
}

// ---------------------------------------------------------------- out proj
__global__ void __launch_bounds__(256) k_gemm_proj(
    const __bf16* __restrict__ A, const __bf16* __restrict__ Wt,
    const float* __restrict__ bias, float* __restrict__ out) {
    const int lane = threadIdx.x & 31;
    const int wave = threadIdx.x >> 5;
    const int hi = lane >> 4, l15 = lane & 15;
    const int tileM = blockIdx.x * 32 + (wave >> 2) * 16;
    const int tileN = blockIdx.y * 256 + (wave & 3) * 64;

    const __bf16* arow = A + (size_t)(tileM + l15) * EMBED;
    const __bf16* bcol[4];
#pragma unroll
    for (int j = 0; j < 4; ++j)
        bcol[j] = Wt + (size_t)(tileN + j * 16 + l15) * EMBED;

    v8f acc[4] = {};
    Frag a0, a1, b0[4], b1[4];
    load_frag(a0, arow, hi);
#pragma unroll
    for (int j = 0; j < 4; ++j) load_frag(b0[j], bcol[j], hi);

#pragma unroll 1
    for (int k = 0; k < EMBED; k += 64) {
        const int k1 = k + 32;
        load_frag(a1, arow + k1, hi);
#pragma unroll
        for (int j = 0; j < 4; ++j) load_frag(b1[j], bcol[j] + k1, hi);
#pragma unroll
        for (int j = 0; j < 4; ++j) acc[j] = wmma_bf16(a0, b0[j], acc[j]);

        const int k2 = (k + 64) & (EMBED - 1);
        load_frag(a0, arow + k2, hi);
#pragma unroll
        for (int j = 0; j < 4; ++j) load_frag(b0[j], bcol[j] + k2, hi);
#pragma unroll
        for (int j = 0; j < 4; ++j) acc[j] = wmma_bf16(a1, b1[j], acc[j]);
    }

#pragma unroll
    for (int j = 0; j < 4; ++j) {
        int col = tileN + j * 16 + l15;
        float bv = bias[col];
#pragma unroll
        for (int r = 0; r < 8; ++r) {
            int row = tileM + r + hi * 8;
            out[(size_t)row * EMBED + col] = acc[j][r] + bv;
        }
    }
}

// ---------------------------------------------------------------- launcher
extern "C" void kernel_launch(void* const* d_in, const int* in_sizes, int n_in,
                              void* d_out, int out_size, void* d_ws, size_t ws_size,
                              hipStream_t stream) {
    const float* query  = (const float*)d_in[0];
    const float* w_qkv  = (const float*)d_in[1];
    const float* b_qkv  = (const float*)d_in[2];
    const float* w_proj = (const float*)d_in[3];
    const float* b_proj = (const float*)d_in[4];
    float* out = (float*)d_out;

    char* ws = (char*)d_ws;
    __bf16* qb     = (__bf16*)(ws + 0);                       // 16 MB [NTOK,EMBED]
    __bf16* wqkvT  = (__bf16*)(ws + (size_t)16777216);        //  6 MB [N3,EMBED]
    __bf16* wprojT = (__bf16*)(ws + (size_t)23068672);        //  2 MB [EMBED,EMBED]
    __bf16* Qb     = (__bf16*)(ws + (size_t)25165824);        // 16 MB [B*H,S,Hd]
    __bf16* Kb     = (__bf16*)(ws + (size_t)41943040);        // 16 MB [B*H,S,Hd]
    __bf16* Vt     = (__bf16*)(ws + (size_t)58720256);        // 16 MB [B*H,Hd,S]
    __bf16* ctx    = (__bf16*)(ws + (size_t)75497472);        // 16 MB [NTOK,EMBED]

    k_cvt<<<(NTOK * EMBED) / 256, 256, 0, stream>>>(query, qb, NTOK * EMBED);
    k_cvt_t<<<(EMBED * N3) / 256, 256, 0, stream>>>(w_qkv, wqkvT, EMBED, N3);
    k_cvt_t<<<(EMBED * EMBED) / 256, 256, 0, stream>>>(w_proj, wprojT, EMBED, EMBED);

    k_gemm_qkv<<<dim3(NTOK / 32, N3 / 256), 256, 0, stream>>>(
        qb, wqkvT, b_qkv, Qb, Kb, Vt);
    k_attn<<<dim3(BATCH * NHEAD, SEQ / 64), 128, 0, stream>>>(Qb, Kb, Vt, ctx);
    k_gemm_proj<<<dim3(NTOK / 32, EMBED / 256), 256, 0, stream>>>(
        ctx, wprojT, b_proj, out);
}
